// token_add_earse_8830452760750
// MI455X (gfx1250) — matrix operand order repaired and checked by reading, not compiled, not run on hardware
//
#include <hip/hip_runtime.h>

// ---------------------------------------------------------------------------
// Problem: B=8, S=2048, D=512, K=8.
// Folded-linear formulation (GELUs only at outermost positions), so the whole
// thing is HBM-bandwidth bound (~70-100MB @ 23.3TB/s).  WMMA bf16 used for the
// logits GEMM (16384x16x512) and the four branch GEMMs.
// ---------------------------------------------------------------------------

typedef __attribute__((ext_vector_type(16))) __bf16 v16bf;
typedef __attribute__((ext_vector_type(8)))  float  v8f;

#define EPSF 1e-5f

__device__ __forceinline__ float gelu_f(float x) {
    return 0.5f * x * (1.0f + erff(x * 0.70710678118654752f));
}

__device__ __forceinline__ unsigned short f2bf(float f) {
    unsigned int u = __float_as_uint(f);
    u += 0x7fffu + ((u >> 16) & 1u);           // round-to-nearest-even
    return (unsigned short)(u >> 16);
}
__device__ __forceinline__ __bf16 tobf(float f) {
    union { unsigned short u; __bf16 h; } c;
    c.u = f2bf(f);
    return c.h;
}

// ---------------------------------------------------------------------------
// 1) Fold W1 = w1a(512x1536) @ w1b(1536x8); row d==512 computes folded bias.
// ---------------------------------------------------------------------------
__global__ void fold_w1_kernel(const float* __restrict__ w1a,
                               const float* __restrict__ b1a,
                               const float* __restrict__ w1b,
                               const float* __restrict__ b1b,
                               float* __restrict__ W1fold,
                               float* __restrict__ b1fold) {
    int idx = blockIdx.x * 256 + threadIdx.x;
    if (idx >= 513 * 8) return;
    int d = idx >> 3, k = idx & 7;
    float acc = 0.f;
    if (d < 512) {
        for (int h = 0; h < 1536; ++h) acc += w1a[(size_t)d * 1536 + h] * w1b[h * 8 + k];
        W1fold[d * 8 + k] = acc;
    } else {
        for (int h = 0; h < 1536; ++h) acc += b1a[h] * w1b[h * 8 + k];
        b1fold[k] = acc + b1b[k];
    }
}

// Pack W1fold into WMMA bf16 B-fragment order: for k-block kb, lane L, elem e:
//   B[K = kb*32 + (L>>4)*16 + e][N = L&15]   (cols >= 8 zero-padded)
// Layout: W1frag[kb*512 + L*16 + e]  -> per-lane contiguous 32B chunk.
__global__ void pack_w1_kernel(const float* __restrict__ W1fold,
                               unsigned short* __restrict__ W1frag) {
    int idx = blockIdx.x * 256 + threadIdx.x;        // < 8192
    if (idx >= 8192) return;
    int e    = idx & 15;
    int lane = (idx >> 4) & 31;
    int kb   = idx >> 9;
    int col  = lane & 15;
    int kk   = kb * 32 + (lane >> 4) * 16 + e;
    float v  = (col < 8) ? W1fold[kk * 8 + col] : 0.f;
    W1frag[idx] = f2bf(v);
}

// ---------------------------------------------------------------------------
// 2) Fold M2 = w2a@w2b (8x8), M3 = w3a@w3b, and their biases.
// ---------------------------------------------------------------------------
__global__ void fold_small_kernel(const float* __restrict__ w2a, const float* __restrict__ b2a,
                                  const float* __restrict__ w2b, const float* __restrict__ b2b,
                                  const float* __restrict__ w3a, const float* __restrict__ b3a,
                                  const float* __restrict__ w3b, const float* __restrict__ b3b,
                                  float* __restrict__ Mf, float* __restrict__ biasf) {
    int idx = threadIdx.x;
    if (idx >= 144) return;
    int br = idx / 72, r = idx % 72;
    const float* wa = br ? w3a : w2a;
    const float* wb = br ? w3b : w2b;
    const float* ba = br ? b3a : b2a;
    const float* bb = br ? b3b : b2b;
    if (r < 64) {
        int j = r >> 3, k = r & 7;
        float acc = 0.f;
        for (int h = 0; h < 1536; ++h) acc += wa[j * 1536 + h] * wb[h * 8 + k];
        Mf[br * 64 + r] = acc;
    } else {
        int k = r - 64;
        float acc = 0.f;
        for (int h = 0; h < 1536; ++h) acc += ba[h] * wb[h * 8 + k];
        biasf[br * 8 + k] = acc + bb[k];
    }
}

// ---------------------------------------------------------------------------
// 3) Logits: one wave per 16 rows of mem.  In-wave LayerNorm, bf16 A-fragments
//    built in contiguous k order (K=half*16+e), 16x v_wmma_f32_16x16x32_bf16 over D,
//    GELU, scatter to sel[B][K][S].
// ---------------------------------------------------------------------------
__global__ __launch_bounds__(32) void logits_kernel(const float* __restrict__ mem,
                                                    const float* __restrict__ g0,
                                                    const float* __restrict__ b0,
                                                    const unsigned short* __restrict__ W1frag,
                                                    const float* __restrict__ b1fold,
                                                    float* __restrict__ sel) {
    __shared__ float gs[512], bs[512], p1[32], p2[32];
    const int lane = threadIdx.x;
    const int tile = blockIdx.x;                     // 0..1023
    for (int i = lane; i < 512; i += 32) { gs[i] = g0[i]; bs[i] = b0[i]; }

    const int row16 = lane & 15;
    const int hf    = lane >> 4;
    const long grow = (long)tile * 16 + row16;
    const float* arow = mem + grow * 512;

    // LN stats: lanes l and l+16 split row l into 256-element halves
    const float4* m4 = (const float4*)(arow + hf * 256);
    float s1 = 0.f, s2 = 0.f;
    for (int i = 0; i < 64; ++i) {
        float4 v = m4[i];
        s1 += v.x + v.y + v.z + v.w;
        s2 += v.x * v.x + v.y * v.y + v.z * v.z + v.w * v.w;
    }
    p1[lane] = s1; p2[lane] = s2;
    __syncthreads();
    float tot1 = p1[row16] + p1[row16 + 16];
    float tot2 = p2[row16] + p2[row16 + 16];
    float mean = tot1 * (1.f / 512.f);
    float var  = tot2 * (1.f / 512.f) - mean * mean;
    float rstd = rsqrtf(var + EPSF);

    v8f acc = {0.f, 0.f, 0.f, 0.f, 0.f, 0.f, 0.f, 0.f};
    for (int kb = 0; kb < 16; ++kb) {
        const int k0 = kb * 32;
        v16bf a;
        #pragma unroll
        for (int g = 0; g < 2; ++g) {                 // e-groups 0..7 and 8..15
            int baseK = k0 + hf * 16 + g * 8;         // A layout: K=half*16+e
            float4 x0 = *(const float4*)(arow + baseK);
            float4 x1 = *(const float4*)(arow + baseK + 4);
            float xs[8] = {x0.x, x0.y, x0.z, x0.w, x1.x, x1.y, x1.z, x1.w};
            #pragma unroll
            for (int j = 0; j < 8; ++j) {
                float v = (xs[j] - mean) * rstd * gs[baseK + j] + bs[baseK + j];
                a[g * 8 + j] = tobf(v);
            }
        }
        union { uint4 u[2]; v16bf v; } wb;
        const uint4* wp = (const uint4*)(W1frag + ((size_t)(kb * 32 + lane)) * 16);
        wb.u[0] = wp[0]; wb.u[1] = wp[1];
        acc = __builtin_amdgcn_wmma_f32_16x16x32_bf16(false, a, false, wb.v,
                                                      (short)0, acc, false, false);
    }

    const int col = row16;                            // C layout: N = lane&15
    if (col < 8) {
        float bias = b1fold[col];
        #pragma unroll
        for (int r = 0; r < 8; ++r) {                 // M = r + half*8
            long orow = (long)tile * 16 + hf * 8 + r;
            int b = (int)(orow >> 11);
            int s = (int)(orow & 2047);
            sel[((size_t)(b * 8 + col)) * 2048 + s] = gelu_f(acc[r] + bias);
        }
    }
}

// ---------------------------------------------------------------------------
// 4) Softmax over S=2048 per (b,k), in place.
// ---------------------------------------------------------------------------
__global__ __launch_bounds__(256) void softmax_kernel(float* __restrict__ sel) {
    __shared__ float red[256];
    int tid = threadIdx.x;
    float* p = sel + (size_t)blockIdx.x * 2048;
    float m = -3.4e38f;
    for (int s = tid; s < 2048; s += 256) m = fmaxf(m, p[s]);
    red[tid] = m; __syncthreads();
    for (int off = 128; off > 0; off >>= 1) {
        if (tid < off) red[tid] = fmaxf(red[tid], red[tid + off]);
        __syncthreads();
    }
    m = red[0]; __syncthreads();
    float sum = 0.f;
    for (int s = tid; s < 2048; s += 256) sum += expf(p[s] - m);
    red[tid] = sum; __syncthreads();
    for (int off = 128; off > 0; off >>= 1) {
        if (tid < off) red[tid] += red[tid + off];
        __syncthreads();
    }
    float inv = 1.f / red[0];
    for (int s = tid; s < 2048; s += 256) p[s] = expf(p[s] - m) * inv;
}

// ---------------------------------------------------------------------------
// 5) LN(control) for both branches: 128 blocks, one per (branch, b, k).
// ---------------------------------------------------------------------------
__global__ __launch_bounds__(256) void ln_control_kernel(const float* __restrict__ control,
                                                         const float* __restrict__ g1,
                                                         const float* __restrict__ b1,
                                                         const float* __restrict__ g2,
                                                         const float* __restrict__ b2,
                                                         float* __restrict__ c_ln) {
    __shared__ float r1[256], r2[256];
    int tid = threadIdx.x;
    int br = blockIdx.x >> 6;
    int rc = blockIdx.x & 63;                         // b*8+k
    const float* row = control + (size_t)rc * 512;
    float x0 = row[tid], x1 = row[tid + 256];
    r1[tid] = x0 + x1; r2[tid] = x0 * x0 + x1 * x1;
    __syncthreads();
    for (int off = 128; off > 0; off >>= 1) {
        if (tid < off) { r1[tid] += r1[tid + off]; r2[tid] += r2[tid + off]; }
        __syncthreads();
    }
    float mean = r1[0] * (1.f / 512.f);
    float var  = r2[0] * (1.f / 512.f) - mean * mean;
    float rstd = rsqrtf(var + EPSF);
    const float* g  = br ? g2 : g1;
    const float* be = br ? b2 : b1;
    float* orow = c_ln + ((size_t)(br * 64 + rc)) * 512;
    orow[tid]       = (x0 - mean) * rstd * g[tid]       + be[tid];
    orow[tid + 256] = (x1 - mean) * rstd * g[tid + 256] + be[tid + 256];
}

// ---------------------------------------------------------------------------
// 6) z[br][b*8+k'][d] = gelu( sum_j c_ln[br][b*8+j][d] * M[br][j][k'] + bias )
// ---------------------------------------------------------------------------
__global__ void branch_z_kernel(const float* __restrict__ c_ln,
                                const float* __restrict__ Mf,
                                const float* __restrict__ biasf,
                                float* __restrict__ z) {
    int idx = blockIdx.x * 256 + threadIdx.x;         // < 65536
    int br  = idx >> 15;
    int rem = idx & 32767;
    int row = rem >> 9;                               // b*8+k'
    int d   = rem & 511;
    int b = row >> 3, kp = row & 7;
    const float* M = Mf + br * 64;
    float acc = biasf[br * 8 + kp];
    #pragma unroll
    for (int j = 0; j < 8; ++j)
        acc += c_ln[((size_t)(br * 64 + b * 8 + j)) * 512 + d] * M[j * 8 + kp];
    z[idx] = gelu_f(acc);
}

// ---------------------------------------------------------------------------
// 7) Generic bf16 WMMA GEMM: one wave per 16x16 C-tile.
//    C[M x N] = act(A[M x Ka] @ Bm[Ka x N] + bias).  Ka % 32 == 0.
// ---------------------------------------------------------------------------
__global__ __launch_bounds__(32) void gemm_bf16_kernel(const float* __restrict__ A, int Ka,
                                                       const float* __restrict__ Bm, int N,
                                                       const float* __restrict__ bias,
                                                       float* __restrict__ C,
                                                       int nNT, int act) {
    const int lane  = threadIdx.x;
    const int tileM = blockIdx.x / nNT;
    const int tileN = blockIdx.x % nNT;
    const int m16 = lane & 15;
    const int hf  = lane >> 4;
    const int col = tileN * 16 + m16;
    const float* arow = A + (size_t)(tileM * 16 + m16) * Ka;

    v8f acc = {0.f, 0.f, 0.f, 0.f, 0.f, 0.f, 0.f, 0.f};
    for (int k0 = 0; k0 < Ka; k0 += 32) {
        v16bf a, b;
        #pragma unroll
        for (int e = 0; e < 16; ++e) {                // A: K=half*16+e
            int kk = k0 + hf * 16 + e;
            a[e] = tobf(arow[kk]);
        }
        #pragma unroll
        for (int e = 0; e < 16; ++e) {                // B: K=half*16+e, N=lane&15
            int kk = k0 + hf * 16 + e;
            b[e] = tobf(Bm[(size_t)kk * N + col]);
        }
        acc = __builtin_amdgcn_wmma_f32_16x16x32_bf16(false, a, false, b,
                                                      (short)0, acc, false, false);
    }
    float bv = bias[col];
    #pragma unroll
    for (int r = 0; r < 8; ++r) {
        int row = tileM * 16 + hf * 8 + r;
        float v = acc[r] + bv;
        if (act) v = gelu_f(v);
        C[(size_t)row * N + col] = v;
    }
}

// ---------------------------------------------------------------------------
// 8) Combine: out = mem * prod_k(1 - sel_k*et_k) + 1 - sum_k(sel_k*at_k)/8.
//    One block per (b, 8 rows of s); et/at staged in LDS (32KB).
// ---------------------------------------------------------------------------
__global__ __launch_bounds__(256) void combine_kernel(const float* __restrict__ mem,
                                                      const float* __restrict__ sel,
                                                      const float* __restrict__ et,
                                                      const float* __restrict__ at,
                                                      float* __restrict__ out) {
    __shared__ float etS[4096];
    __shared__ float atS[4096];
    int tid = threadIdx.x;
    int b   = blockIdx.x >> 8;
    int s0  = (blockIdx.x & 255) * 8;
    const float* etb = et + (size_t)b * 4096;
    const float* atb = at + (size_t)b * 4096;
    for (int i = tid; i < 4096; i += 256) { etS[i] = etb[i]; atS[i] = atb[i]; }
    __syncthreads();
    int d0 = tid * 2;
    for (int r = 0; r < 8; ++r) {
        int s = s0 + r;
        float sl[8];
        #pragma unroll
        for (int k = 0; k < 8; ++k) sl[k] = sel[((size_t)(b * 8 + k)) * 2048 + s];
        size_t base = ((size_t)(b * 2048 + s)) * 512 + d0;
        float2 mv = *(const float2*)(mem + base);
        float mvv[2] = {mv.x, mv.y};
        float res[2];
        #pragma unroll
        for (int j = 0; j < 2; ++j) {
            int d = d0 + j;
            float prod = 1.f, dot = 0.f;
            #pragma unroll
            for (int k = 0; k < 8; ++k) {
                float se = sl[k];
                prod *= (1.f - se * etS[k * 512 + d]);
                dot  += se * atS[k * 512 + d];
            }
            res[j] = mvv[j] * prod + 1.f - dot * 0.125f;
        }
        float2 ov = {res[0], res[1]};
        *(float2*)(out + base) = ov;
    }
}

// ---------------------------------------------------------------------------
// Host launch
// ---------------------------------------------------------------------------
extern "C" void kernel_launch(void* const* d_in, const int* in_sizes, int n_in,
                              void* d_out, int out_size, void* d_ws, size_t ws_size,
                              hipStream_t stream) {
    const float* mem    = (const float*)d_in[0];
    const float* control= (const float*)d_in[1];
    const float* ln0_g  = (const float*)d_in[2];
    const float* ln0_b  = (const float*)d_in[3];
    const float* w1a    = (const float*)d_in[4];
    const float* b1a    = (const float*)d_in[5];
    const float* w1b    = (const float*)d_in[6];
    const float* b1b    = (const float*)d_in[7];
    const float* ln1_g  = (const float*)d_in[8];
    const float* ln1_b  = (const float*)d_in[9];
    const float* ln2_g  = (const float*)d_in[10];
    const float* ln2_b  = (const float*)d_in[11];
    const float* w2a    = (const float*)d_in[12];
    const float* b2a    = (const float*)d_in[13];
    const float* w2b    = (const float*)d_in[14];
    const float* b2b    = (const float*)d_in[15];
    const float* w2c    = (const float*)d_in[16];
    const float* b2c    = (const float*)d_in[17];
    const float* w2d    = (const float*)d_in[18];
    const float* b2d    = (const float*)d_in[19];
    const float* w3a    = (const float*)d_in[20];
    const float* b3a    = (const float*)d_in[21];
    const float* w3b    = (const float*)d_in[22];
    const float* b3b    = (const float*)d_in[23];
    const float* w3c    = (const float*)d_in[24];
    const float* b3c    = (const float*)d_in[25];
    const float* w3d    = (const float*)d_in[26];
    const float* b3d    = (const float*)d_in[27];
    float* out = (float*)d_out;
    (void)in_sizes; (void)n_in; (void)out_size; (void)ws_size;

    char* ws = (char*)d_ws;
    size_t off = 0;
    auto carve = [&](size_t bytes) -> char* {
        char* p = ws + off;
        off += (bytes + 255) & ~size_t(255);
        return p;
    };
    float*          W1fold = (float*)carve(512 * 8 * 4);
    float*          b1fold = (float*)carve(8 * 4);
    unsigned short* W1frag = (unsigned short*)carve(8192 * 2);
    float*          sel    = (float*)carve(64 * 2048 * 4);
    float*          c_ln   = (float*)carve(2 * 64 * 512 * 4);
    float*          Mf     = (float*)carve(2 * 64 * 4);
    float*          biasf  = (float*)carve(2 * 8 * 4);
    float*          z      = (float*)carve(2 * 64 * 512 * 4);
    float*          H      = (float*)carve(2 * 64 * 1536 * 4);
    float*          et     = (float*)carve(64 * 512 * 4);
    float*          at     = (float*)carve(64 * 512 * 4);

    fold_w1_kernel   <<<17,   256, 0, stream>>>(w1a, b1a, w1b, b1b, W1fold, b1fold);
    pack_w1_kernel   <<<32,   256, 0, stream>>>(W1fold, W1frag);
    fold_small_kernel<<<1,    256, 0, stream>>>(w2a, b2a, w2b, b2b, w3a, b3a, w3b, b3b, Mf, biasf);
    logits_kernel    <<<1024, 32,  0, stream>>>(mem, ln0_g, ln0_b, W1frag, b1fold, sel);
    softmax_kernel   <<<64,   256, 0, stream>>>(sel);
    ln_control_kernel<<<128,  256, 0, stream>>>(control, ln1_g, ln1_b, ln2_g, ln2_b, c_ln);
    branch_z_kernel  <<<256,  256, 0, stream>>>(c_ln, Mf, biasf, z);
    // branch 2 (erase): H2 = z2@w2c + b2c ; et = gelu(H2@w2d + b2d)
    gemm_bf16_kernel <<<4 * 96, 32, 0, stream>>>(z,             512,  w2c, 1536, b2c, H,             96, 0);
    gemm_bf16_kernel <<<4 * 32, 32, 0, stream>>>(H,             1536, w2d, 512,  b2d, et,            32, 1);
    // branch 3 (add):   H3 = z3@w3c + b3c ; at = gelu(H3@w3d + b3d)
    gemm_bf16_kernel <<<4 * 96, 32, 0, stream>>>(z + 64 * 512,  512,  w3c, 1536, b3c, H + 64 * 1536, 96, 0);
    gemm_bf16_kernel <<<4 * 32, 32, 0, stream>>>(H + 64 * 1536, 1536, w3d, 512,  b3d, at,            32, 1);
    combine_kernel   <<<2048, 256, 0, stream>>>(mem, sel, et, at, out);
}
